// FlashRWLargeAttention_51410758533419
// MI455X (gfx1250) — matrix operand order repaired and checked by reading, not compiled, not run on hardware
//
#include <hip/hip_runtime.h>
#include <hip/hip_bf16.h>

// FlashRW attention block for MI455X (gfx1250, wave32, bf16 WMMA, f32 accum).
// Pipeline:
//   [cvt hidden->bf16] [transpose+cvt w_qkv, w_dense -> bf16 N-major]
//   -> [bf16 GEMM qkv (WMMA, double-buffered async->LDS)] -> [RoPE scatter]
//   -> [flash attention (WMMA)] -> [bf16 GEMM dense (WMMA)]
// GEMM staging uses gfx1250 global_load_async_to_lds_b128 (ASYNCcnt) with
// 2-deep LDS double buffering: next tile's DMA overlaps current tile's WMMAs;
// s_wait_asynccnt 8 retires only the previous tile (8 in-order copies/thread).
//
// Workspace (~64 MB peak; attn-bf16 aliases the dead qkv-f32 region):
//   qkv     f32  [2048][3072]  @ 0          (25165824 B)   } aliased later by
//   attnb   bf16 [2048][2048]  @ 0          ( 8388608 B)   } attn output
//   hb      bf16 [2048][2048]  @ 25165824   ( 8388608 B)
//   wqkvT   bf16 [3072][2048]  @ 33554432   (12582912 B)
//   wdenseT bf16 [2048][2048]  @ 46137344   ( 8388608 B)
//   Qb      bf16 [32][2048][64]@ 54525952   ( 8388608 B)
//   Kb      bf16 [ 8][2048][64]@ 62914560   ( 2097152 B)
//   Vt      bf16 [ 8][64][2048]@ 65011712   ( 2097152 B)   (V d-major)

typedef __attribute__((ext_vector_type(16))) __bf16 v16bf;
typedef __attribute__((ext_vector_type(8)))  __bf16 v8bf;
typedef __attribute__((ext_vector_type(8)))  float  v8f;
typedef __attribute__((ext_vector_type(4)))  int    v4i;

static_assert(sizeof(__bf16) == 2, "bf16 size");

#define DEV_INLINE __device__ __forceinline__

// ---- gfx1250 async global->LDS (param types confirmed by compile probe) ----
#if __has_builtin(__builtin_amdgcn_global_load_async_to_lds_b128) && \
    __has_builtin(__builtin_amdgcn_s_wait_asynccnt)
#define USE_ASYNC_LDS 1
DEV_INLINE void async_copy16(const void* g, void* l) {
  // Param 0: int4* in AS(1) ("__device__"), param 1: int4* in AS(3) ("__shared__").
  // Pointers built via inttoptr: flat LDS addr low 32 bits == LDS offset.
  __builtin_amdgcn_global_load_async_to_lds_b128(
      (__attribute__((address_space(1))) v4i*)(unsigned long long)(uintptr_t)g,
      (__attribute__((address_space(3))) v4i*)(unsigned int)(uintptr_t)l,
      0, 0);
}
#endif

// ---------------------------------------------------------------------------
// elementwise f32 -> bf16 (hidden states)
// ---------------------------------------------------------------------------
__global__ __launch_bounds__(256) void convert_bf16(const float* __restrict__ src,
                                                    __bf16* __restrict__ dst) {
  int i = (blockIdx.x * 256 + threadIdx.x) * 4;
  const float4 f = *(const float4*)(src + i);
  dst[i + 0] = (__bf16)f.x;
  dst[i + 1] = (__bf16)f.y;
  dst[i + 2] = (__bf16)f.z;
  dst[i + 3] = (__bf16)f.w;
}

// ---------------------------------------------------------------------------
// W f32 [K][N] -> Wt bf16 [N][K] (64x64 LDS tile transpose, coalesced both ways)
// ---------------------------------------------------------------------------
__global__ __launch_bounds__(256) void transpose_f32_bf16(const float* __restrict__ W,
                                                          __bf16* __restrict__ Wt,
                                                          int K, int N) {
  __shared__ float tile[64][65];
  const int k0 = blockIdx.y * 64, n0 = blockIdx.x * 64;
  const int tx = threadIdx.x & 63, ty = threadIdx.x >> 6;  // 64 x 4
#pragma unroll
  for (int rr = 0; rr < 64; rr += 4)
    tile[rr + ty][tx] = W[(size_t)(k0 + rr + ty) * N + n0 + tx];
  __syncthreads();
#pragma unroll
  for (int rr = 0; rr < 64; rr += 4)
    Wt[(size_t)(n0 + rr + ty) * K + k0 + tx] = (__bf16)tile[tx][rr + ty];
}

// ---------------------------------------------------------------------------
// GEMM: C[M][N] = A[M][K] @ B[K][N] with A bf16 row-major, Bt bf16 = B^T
// (N-major), C f32. 256 threads / 8 waves, 128x128 tile, BK=64; each wave does
// 2x4 v_wmma_f32_16x16x32_bf16 tiles. Double-buffered async->LDS staging.
// ---------------------------------------------------------------------------
constexpr int GBM = 128, GBN = 128, GBK = 64, GLDT = GBK + 8;  // pad keeps rows 16B-aligned

#ifdef USE_ASYNC_LDS
DEV_INLINE void stage_tile_async(const __bf16* __restrict__ A, const __bf16* __restrict__ Bt,
                                 __bf16* As, __bf16* Bs,
                                 int m0, int n0, int K, int k0, int tid) {
#pragma unroll
  for (int t = 0; t < 4; ++t) {
    int c = tid + t * 256, r = c >> 3, c8 = (c & 7) * 8;
    async_copy16(A  + (size_t)(m0 + r) * K + k0 + c8, &As[r * GLDT + c8]);
    async_copy16(Bt + (size_t)(n0 + r) * K + k0 + c8, &Bs[r * GLDT + c8]);
  }
}
#endif

__global__ __launch_bounds__(256) void gemm_bf16_tt(const __bf16* __restrict__ A,
                                                    const __bf16* __restrict__ Bt,
                                                    float* __restrict__ C,
                                                    int M, int N, int K) {
  __shared__ alignas(16) __bf16 As[2][GBM * GLDT];
  __shared__ alignas(16) __bf16 Bs[2][GBN * GLDT];

  const int tid  = threadIdx.x;
  const int lane = tid & 31;
  const int wid  = tid >> 5;
  const int half = lane >> 4;
  const int l16  = lane & 15;
  const int wm   = wid & 3;   // 32-row slab
  const int wn   = wid >> 2;  // 64-col slab
  const int m0   = blockIdx.y * GBM;
  const int n0   = blockIdx.x * GBN;

  const v8f vzero = {0.f, 0.f, 0.f, 0.f, 0.f, 0.f, 0.f, 0.f};
  v8f acc[2][4];
#pragma unroll
  for (int i = 0; i < 2; ++i)
#pragma unroll
    for (int j = 0; j < 4; ++j) acc[i][j] = vzero;

#ifdef USE_ASYNC_LDS
  stage_tile_async(A, Bt, As[0], Bs[0], m0, n0, K, 0, tid);  // prologue: 8 copies in flight
#endif

  int buf = 0;
  for (int k0 = 0; k0 < K; k0 += GBK, buf ^= 1) {
#ifdef USE_ASYNC_LDS
    if (k0 + GBK < K) {
      // issue next tile's DMA, then retire only the previous tile's 8 copies:
      // async loads complete in order, so ASYNCcnt <= 8 means tile `buf` landed.
      stage_tile_async(A, Bt, As[buf ^ 1], Bs[buf ^ 1], m0, n0, K, k0 + GBK, tid);
      __builtin_amdgcn_s_wait_asynccnt(8);
    } else {
      __builtin_amdgcn_s_wait_asynccnt(0);
    }
#else
    uint4 ta[4], tb[4];
#pragma unroll
    for (int t = 0; t < 4; ++t) {
      int c = tid + t * 256, r = c >> 3, c8 = (c & 7) * 8;
      ta[t] = *(const uint4*)(A  + (size_t)(m0 + r) * K + k0 + c8);
      tb[t] = *(const uint4*)(Bt + (size_t)(n0 + r) * K + k0 + c8);
    }
#pragma unroll
    for (int t = 0; t < 4; ++t) {
      int c = tid + t * 256, r = c >> 3, c8 = (c & 7) * 8;
      *(uint4*)&As[buf][r * GLDT + c8] = ta[t];
      *(uint4*)&Bs[buf][r * GLDT + c8] = tb[t];
    }
#endif
    __syncthreads();

    const __bf16* Ab = As[buf];
    const __bf16* Bb = Bs[buf];
#pragma unroll
    for (int kc = 0; kc < GBK; kc += 32) {
      v16bf a[2], b[4];
#pragma unroll
      for (int i = 0; i < 2; ++i) {
        // A frag (16x32 bf16): lane row = l16; K = kc+{0..7,16..23} / {8..15,24..31}
        const __bf16* p = &Ab[(wm * 32 + i * 16 + l16) * GLDT + kc + half * 8];
        v8bf lo = *(const v8bf*)p;
        v8bf hi = *(const v8bf*)(p + 16);
#pragma unroll
        for (int e = 0; e < 8; ++e) { a[i][e] = lo[e]; a[i][e + 8] = hi[e]; }
      }
#pragma unroll
      for (int j = 0; j < 4; ++j) {
        // B frag (32x16): lane col = l16; K = kc+{0..7,16..23} / {8..15,24..31}
        const __bf16* p = &Bb[(wn * 64 + j * 16 + l16) * GLDT + kc + half * 8];
        v8bf lo = *(const v8bf*)p;
        v8bf hi = *(const v8bf*)(p + 16);
#pragma unroll
        for (int e = 0; e < 8; ++e) { b[j][e] = lo[e]; b[j][e + 8] = hi[e]; }
      }
#pragma unroll
      for (int i = 0; i < 2; ++i)
#pragma unroll
        for (int j = 0; j < 4; ++j)
          acc[i][j] = __builtin_amdgcn_wmma_f32_16x16x32_bf16(
              false, a[i], false, b[j], (short)0, acc[i][j], false, false);
    }
    __syncthreads();  // all waves done reading buf before it is re-staged
  }

  // D layout: vgpr r -> row r + 8*half ; col = l16
#pragma unroll
  for (int i = 0; i < 2; ++i)
#pragma unroll
    for (int j = 0; j < 4; ++j) {
      int row = m0 + wm * 32 + i * 16 + 8 * half;
      int col = n0 + wn * 64 + j * 16 + l16;
#pragma unroll
      for (int r = 0; r < 8; ++r)
        C[(size_t)(row + r) * N + col] = acc[i][j][r];
    }
}

// ---------------------------------------------------------------------------
// RoPE + scatter: qkv f32 (S,G,6,D) -> Qb bf16 [32][S][64] (pre-scaled D^-1/2),
// Kb bf16 [8][S][64], Vt bf16 [8][64][S].
// ---------------------------------------------------------------------------
__global__ __launch_bounds__(256) void rope_scatter(const float* __restrict__ qkv,
                                                    const float* __restrict__ cosb,
                                                    const float* __restrict__ sinb,
                                                    __bf16* __restrict__ Qb,
                                                    __bf16* __restrict__ Kb,
                                                    __bf16* __restrict__ Vt) {
  constexpr int S = 2048;
  int idx = blockIdx.x * 256 + threadIdx.x;  // over S*8*6*64
  int d  = idx & 63;
  int t  = idx >> 6;
  int hh = t % 6;  t /= 6;
  int g  = t & 7;
  int s  = t >> 3;

  float x = qkv[idx];
  float val = x;
  if (hh < 5) {  // rotary on q heads and k
    int  p  = d & 31;
    bool hi = d >= 32;
    float c  = cosb[s * 32 + p];
    float sn = sinb[s * 32 + p];
    float other = qkv[idx + (hi ? -32 : 32)];
    val = hi ? (x * c + other * sn) : (x * c - other * sn);
  }
  if (hh < 4) {
    int head = g * 4 + hh;
    Qb[((size_t)head * S + s) * 64 + d] = (__bf16)(val * 0.125f);  // SCALE = 64^-0.5
  } else if (hh == 4) {
    Kb[((size_t)g * S + s) * 64 + d] = (__bf16)val;
  } else {
    Vt[((size_t)g * 64 + d) * S + s] = (__bf16)val;
  }
}

// ---------------------------------------------------------------------------
// Flash attention (causal, GQA): grid (S/128, 32 heads), 256 threads.
// Wave w owns 16 q rows; Bc=64. QK^T and PV via bf16 WMMA, online softmax in
// D-layout (row = vgpr + 8*half, col = lane&15), shfl reductions within the
// 16-lane halves, P transposed through a per-wave LDS slab. bf16 output.
// ---------------------------------------------------------------------------
__global__ __launch_bounds__(256) void flash_attn(const __bf16* __restrict__ Qb,
                                                  const __bf16* __restrict__ Kb,
                                                  const __bf16* __restrict__ Vt,
                                                  __bf16* __restrict__ attnb) {
  constexpr int S = 2048, D = 64;
  __shared__ alignas(16) __bf16 Pl[8][16][64];

  const int tid  = threadIdx.x;
  const int lane = tid & 31;
  const int wid  = tid >> 5;
  const int half = lane >> 4;
  const int l16  = lane & 15;
  const int head = blockIdx.y;
  const int g    = head >> 2;
  const int q0   = blockIdx.x * 128 + wid * 16;

  const __bf16* qrow = Qb + ((size_t)head * S + q0 + l16) * D;
  v16bf qa[2];
#pragma unroll
  for (int c = 0; c < 2; ++c) {
    v8bf lo = *(const v8bf*)(qrow + c * 32 + half * 8);
    v8bf hi = *(const v8bf*)(qrow + c * 32 + 16 + half * 8);
#pragma unroll
    for (int e = 0; e < 8; ++e) { qa[c][e] = lo[e]; qa[c][e + 8] = hi[e]; }
  }

  const v8f vzero = {0.f, 0.f, 0.f, 0.f, 0.f, 0.f, 0.f, 0.f};
  float mrow[8], lrow[8];
  v8f o[4];
#pragma unroll
  for (int r = 0; r < 8; ++r) { mrow[r] = -3.0e38f; lrow[r] = 0.f; }
#pragma unroll
  for (int dt = 0; dt < 4; ++dt) o[dt] = vzero;

  const int jmax = (q0 + 15) >> 6;
  for (int j = 0; j <= jmax; ++j) {
    const int t0 = j * 64;

    v8f sc[4];
#pragma unroll
    for (int nt = 0; nt < 4; ++nt) sc[nt] = vzero;
#pragma unroll
    for (int nt = 0; nt < 4; ++nt)
#pragma unroll
      for (int c = 0; c < 2; ++c) {
        const __bf16* kp = Kb + ((size_t)g * S + t0 + nt * 16 + l16) * D + c * 32 + half * 8;
        v8bf lo = *(const v8bf*)kp;
        v8bf hi = *(const v8bf*)(kp + 16);
        v16bf b;
#pragma unroll
        for (int e = 0; e < 8; ++e) { b[e] = lo[e]; b[e + 8] = hi[e]; }
        sc[nt] = __builtin_amdgcn_wmma_f32_16x16x32_bf16(
            false, qa[c], false, b, (short)0, sc[nt], false, false);
      }

#pragma unroll
    for (int r = 0; r < 8; ++r) {
      const int qidx = q0 + r + 8 * half;
      float mx = -3.0e38f;
#pragma unroll
      for (int nt = 0; nt < 4; ++nt) {
        int col = t0 + nt * 16 + l16;
        float v = sc[nt][r];
        if (col > qidx) { v = -1.0e30f; sc[nt][r] = v; }
        mx = fmaxf(mx, v);
      }
#pragma unroll
      for (int off = 8; off >= 1; off >>= 1)
        mx = fmaxf(mx, __shfl_xor(mx, off, 32));
      const float mnew  = fmaxf(mrow[r], mx);
      const float alpha = __expf(mrow[r] - mnew);
      lrow[r] *= alpha;
#pragma unroll
      for (int dt = 0; dt < 4; ++dt) o[dt][r] *= alpha;
      float rsum = 0.f;
#pragma unroll
      for (int nt = 0; nt < 4; ++nt) {
        float p = __expf(sc[nt][r] - mnew);
        sc[nt][r] = p;
        rsum += p;
      }
#pragma unroll
      for (int off = 8; off >= 1; off >>= 1)
        rsum += __shfl_xor(rsum, off, 32);
      lrow[r] += rsum;
      mrow[r]  = mnew;
    }

    // P: D-layout -> A-layout via per-wave LDS slab (wave-synchronous)
#pragma unroll
    for (int nt = 0; nt < 4; ++nt)
#pragma unroll
      for (int r = 0; r < 8; ++r)
        Pl[wid][r + 8 * half][nt * 16 + l16] = (__bf16)sc[nt][r];

    v16bf pa[2];
#pragma unroll
    for (int c = 0; c < 2; ++c) {
      const __bf16* pp = &Pl[wid][l16][c * 32 + half * 8];
      v8bf lo = *(const v8bf*)pp;
      v8bf hi = *(const v8bf*)(pp + 16);
#pragma unroll
      for (int e = 0; e < 8; ++e) { pa[c][e] = lo[e]; pa[c][e + 8] = hi[e]; }
    }

#pragma unroll
    for (int dt = 0; dt < 4; ++dt)
#pragma unroll
      for (int c = 0; c < 2; ++c) {
        const __bf16* vp = Vt + ((size_t)g * D + dt * 16 + l16) * S + t0 + c * 32 + half * 8;
        v8bf lo = *(const v8bf*)vp;
        v8bf hi = *(const v8bf*)(vp + 16);
        v16bf b;
#pragma unroll
        for (int e = 0; e < 8; ++e) { b[e] = lo[e]; b[e + 8] = hi[e]; }
        o[dt] = __builtin_amdgcn_wmma_f32_16x16x32_bf16(
            false, pa[c], false, b, (short)0, o[dt], false, false);
      }
  }

  // epilogue: attnb[s][head*64 + d] = (bf16)(O / l)  -- bf16 feeds dense GEMM
#pragma unroll
  for (int dt = 0; dt < 4; ++dt)
#pragma unroll
    for (int r = 0; r < 8; ++r)
      attnb[(size_t)(q0 + r + 8 * half) * 2048 + head * 64 + dt * 16 + l16] =
          (__bf16)(o[dt][r] / lrow[r]);
}

// ---------------------------------------------------------------------------
extern "C" void kernel_launch(void* const* d_in, const int* in_sizes, int n_in,
                              void* d_out, int out_size, void* d_ws, size_t ws_size,
                              hipStream_t stream) {
  (void)in_sizes; (void)n_in; (void)out_size; (void)ws_size;
  const float* hidden  = (const float*)d_in[0];
  const float* cosb    = (const float*)d_in[1];
  const float* sinb    = (const float*)d_in[2];
  const float* w_qkv   = (const float*)d_in[3];
  const float* w_dense = (const float*)d_in[4];
  float* out = (float*)d_out;

  char* ws = (char*)d_ws;
  float*  qkv     = (float*)(ws);                // 2048*3072 f32  (dead after rope)
  __bf16* attnb   = (__bf16*)(ws);               // aliases qkv region
  __bf16* hb      = (__bf16*)(ws + 25165824);    // 2048*2048
  __bf16* wqkvT   = (__bf16*)(ws + 33554432);    // 3072*2048
  __bf16* wdenseT = (__bf16*)(ws + 46137344);    // 2048*2048
  __bf16* Qb      = (__bf16*)(ws + 54525952);    // 32*2048*64
  __bf16* Kb      = (__bf16*)(ws + 62914560);    // 8*2048*64
  __bf16* Vt      = (__bf16*)(ws + 65011712);    // 8*64*2048

  // 0) one-time precision/layout conversion
  convert_bf16<<<(2048 * 2048) / 1024, 256, 0, stream>>>(hidden, hb);
  transpose_f32_bf16<<<dim3(3072 / 64, 2048 / 64), 256, 0, stream>>>(w_qkv, wqkvT, 2048, 3072);
  transpose_f32_bf16<<<dim3(2048 / 64, 2048 / 64), 256, 0, stream>>>(w_dense, wdenseT, 2048, 2048);

  // 1) qkv = hidden @ w_qkv   (M=2048, N=3072, K=2048)
  gemm_bf16_tt<<<dim3(3072 / 128, 2048 / 128), 256, 0, stream>>>(
      hb, wqkvT, qkv, 2048, 3072, 2048);

  // 2) RoPE + scatter
  rope_scatter<<<(2048 * 3072) / 256, 256, 0, stream>>>(qkv, cosb, sinb, Qb, Kb, Vt);

  // 3) causal GQA flash attention (writes bf16)
  flash_attn<<<dim3(2048 / 128, 32), 256, 0, stream>>>(Qb, Kb, Vt, attnb);

  // 4) out = attn @ w_dense   (M=2048, N=2048, K=2048)
  gemm_bf16_tt<<<dim3(2048 / 128, 2048 / 128), 256, 0, stream>>>(
      attnb, wdenseT, out, 2048, 2048, 2048);
}
